// GroupQueryMultiHeadAttention_25735444037650
// MI455X (gfx1250) — matrix operand order repaired and checked by reading, not compiled, not run on hardware
//
#include <hip/hip_runtime.h>
#include <hip/hip_bf16.h>

// ---------------------------------------------------------------------------
// GQA attention for MI455X (gfx1250): bf16 WMMA GEMMs + flash attention.
// D=1024, H=16, G=4, B=2, S=1024, dh=64, SG=4096.
// Pipeline: one-time f32->bf16 conversion (weights transposed to [N][K]),
// then all GEMMs run bf16 tiles staged by the Tensor Data Mover (TDM) when
// available, consumed by v_wmma_f32_16x16x32_bf16.
// ---------------------------------------------------------------------------

typedef __attribute__((ext_vector_type(16))) __bf16 v16bf;
typedef __attribute__((ext_vector_type(8)))  float  v8f;

union Frag16 { v16bf v; uint4 q[2]; };
union U4S8  { uint4 u; unsigned short s[8]; };

__device__ __forceinline__ unsigned short f2bf(float f) {
    unsigned int u = __float_as_uint(f);
    u += 0x7FFFu + ((u >> 16) & 1u);      // round-to-nearest-even
    return (unsigned short)(u >> 16);
}

__device__ __forceinline__ v8f vzero8() {
    v8f z = {0.f,0.f,0.f,0.f,0.f,0.f,0.f,0.f};
    return z;
}

#define HAS_TDM __has_builtin(__builtin_amdgcn_tensor_load_to_lds)

#if HAS_TDM
typedef __attribute__((ext_vector_type(4))) unsigned int v4u;
typedef __attribute__((ext_vector_type(8))) int          v8i;
typedef __attribute__((ext_vector_type(4))) int          v4i;

// Issue one TDM load of a [tile_rows x 32] bf16 tile (row length 32 elems)
// from a row-major [nrows x row_len] bf16 tensor into LDS. pad_enable adds
// 4 DWORDs (8 ushorts) after every 16 DWORDs (one 32-elem row), producing a
// padded LDS row stride of 40 ushorts (80 B, 16B-aligned fragment runs).
__device__ __forceinline__ void tdm_load_tile_bf16(
    unsigned int lds_addr, const unsigned short* gptr,
    unsigned int row_len, unsigned int nrows, unsigned int tile_rows)
{
    unsigned long long ga = (unsigned long long)(size_t)gptr;
    v4u g0 = {
        1u,                                       // count=1 (valid user D#)
        lds_addr,                                 // LDS byte address
        (unsigned int)ga,                         // global_addr[31:0]
        ((unsigned int)(ga >> 32) & 0x01FFFFFFu) | 0x80000000u  // addr[56:32] | type=2
    };
    v8i g1 = {
        // data_size=2B(1) | pad_enable | pad_interval=16dw(3) | pad_amount=4dw(3)
        (int)((1u << 16) | (1u << 20) | (3u << 22) | (3u << 25)),
        (int)((row_len & 0xFFFFu) << 16),                 // tensor_dim0[15:0] @ [63:48]
        (int)((row_len >> 16) | ((nrows & 0xFFFFu) << 16)),// td0[31:16] | td1[15:0]
        (int)((nrows >> 16) | (32u << 16)),               // td1[31:16] | tile_dim0=32
        (int)(tile_rows & 0xFFFFu),                       // tile_dim1 | tile_dim2=0
        (int)row_len,                                     // tensor_dim0_stride[31:0]
        0, 0
    };
    v4i z4 = {0, 0, 0, 0};
    v8i z8 = {0, 0, 0, 0, 0, 0, 0, 0};
    // 6-arg (clang-23 / therock) form: (g0, g1, g2, g3, g4, cpol)
    __builtin_amdgcn_tensor_load_to_lds(g0, g1, z4, z4, z8, 0);
}
#endif

// ---------------------------------------------------------------------------
// One-time conversion kernels.
// ---------------------------------------------------------------------------
__global__ __launch_bounds__(256) void cvt_bf16(
    const float* __restrict__ src, unsigned short* __restrict__ dst)
{
    int i = (blockIdx.x * 256 + threadIdx.x) * 4;
    float4 v = *(const float4*)(src + i);
    unsigned int lo = (unsigned int)f2bf(v.x) | ((unsigned int)f2bf(v.y) << 16);
    unsigned int hi = (unsigned int)f2bf(v.z) | ((unsigned int)f2bf(v.w) << 16);
    uint2 p; p.x = lo; p.y = hi;
    *(uint2*)(dst + i) = p;
}

// dst[c*R + r] = bf16(src[r*C + c])   (weights -> transposed bf16 [N][K])
__global__ __launch_bounds__(256) void transpose_cvt_bf16(
    const float* __restrict__ src, unsigned short* __restrict__ dst, int R, int C)
{
    __shared__ float tile[32][33];
    const int r0 = blockIdx.y * 32, c0 = blockIdx.x * 32;
    const int tx = threadIdx.x & 31, ty = threadIdx.x >> 5;   // 32 x 8
    #pragma unroll
    for (int i = 0; i < 32; i += 8)
        tile[ty + i][tx] = src[(size_t)(r0 + ty + i) * C + c0 + tx];
    __syncthreads();
    #pragma unroll
    for (int i = 0; i < 32; i += 8)
        dst[(size_t)(c0 + ty + i) * R + r0 + tx] = f2bf(tile[tx][ty + i]);
}

// ---------------------------------------------------------------------------
// Tiled GEMM:  C[M,N] = A[M,K] * Bt[N,K]^T + bias[N]   (A, Bt bf16 row-major)
// Block tile 128x64, BK=32. 256 threads = 8 waves (4 M x 2 N), each wave
// computes a 32x32 sub-tile as 2x2 WMMA 16x16x32 bf16 tiles.
// Tiles staged to LDS by TDM (padded by descriptor) or b128 copies (fallback).
// ---------------------------------------------------------------------------
template<bool OUT_BF16>
__global__ __launch_bounds__(256) void gemm_wmma(
    const unsigned short* __restrict__ A, const unsigned short* __restrict__ Bt,
    const float* __restrict__ bias, void* __restrict__ Cout,
    int M, int N, int K)
{
    constexpr int BM = 128, BN = 64, BK = 32;
    constexpr int LDS_A = 40, LDS_B = 40;      // padded stride (16B-aligned runs)
    __shared__ __align__(16) unsigned short As[BM * LDS_A];
    __shared__ __align__(16) unsigned short Bs[BN * LDS_B];

    const int tid  = threadIdx.x;
    const int m0   = blockIdx.y * BM;
    const int n0   = blockIdx.x * BN;
    const int w    = tid >> 5;
    const int lane = tid & 31;
    const int l16  = lane & 15;
    const int lh   = lane >> 4;
    const int wm   = w & 3;          // 0..3 -> M sub-block
    const int wn   = w >> 2;         // 0..1 -> N sub-block

    v8f c[2][2];
    #pragma unroll
    for (int i = 0; i < 2; ++i)
        #pragma unroll
        for (int j = 0; j < 2; ++j) c[i][j] = vzero8();

    for (int kt = 0; kt < K; kt += BK) {
#if HAS_TDM
        {
            int wfl = __builtin_amdgcn_readfirstlane(w);   // scalar branch: TDM ignores EXEC
            if (wfl == 0) {
                tdm_load_tile_bf16((unsigned int)(size_t)(void*)As,
                                   A + (size_t)m0 * K + kt, (unsigned)K, (unsigned)M, BM);
                __builtin_amdgcn_s_wait_tensorcnt(0);
            } else if (wfl == 1) {
                tdm_load_tile_bf16((unsigned int)(size_t)(void*)Bs,
                                   Bt + (size_t)n0 * K + kt, (unsigned)K, (unsigned)N, BN);
                __builtin_amdgcn_s_wait_tensorcnt(0);
            }
        }
        __syncthreads();
#else
        // --- fallback: vector-copy staging (bf16, already padded stride) ---
        #pragma unroll
        for (int i = 0; i < 2; ++i) {
            int idx = tid + i * 256;               // 512 x uint4 for A (128x32)
            int row = idx >> 2, c8 = (idx & 3) << 3;
            *(uint4*)&As[row * LDS_A + c8] =
                *(const uint4*)(A + (size_t)(m0 + row) * K + kt + c8);
        }
        {
            int row = tid >> 2, c8 = (tid & 3) << 3;  // 256 x uint4 for B (64x32)
            *(uint4*)&Bs[row * LDS_B + c8] =
                *(const uint4*)(Bt + (size_t)(n0 + row) * K + kt + c8);
        }
        __syncthreads();
#endif

        // --- fragments per ISA layouts ---
        Frag16 fa[2], fb[2];
        const int k0a = lh * 8;
        #pragma unroll
        for (int mt = 0; mt < 2; ++mt) {
            const unsigned short* pr = &As[(wm * 32 + mt * 16 + l16) * LDS_A];
            fa[mt].q[0] = *(const uint4*)(pr + k0a);
            fa[mt].q[1] = *(const uint4*)(pr + 16 + k0a);
        }
        #pragma unroll
        for (int nt = 0; nt < 2; ++nt) {
            const unsigned short* pb = &Bs[(wn * 32 + nt * 16 + l16) * LDS_B + lh * 8];
            fb[nt].q[0] = *(const uint4*)(pb);
            fb[nt].q[1] = *(const uint4*)(pb + 16);
        }
        #pragma unroll
        for (int mt = 0; mt < 2; ++mt)
            #pragma unroll
            for (int nt = 0; nt < 2; ++nt)
                c[mt][nt] = __builtin_amdgcn_wmma_f32_16x16x32_bf16(
                    false, fa[mt].v, false, fb[nt].v, (short)0, c[mt][nt], false, false);
        __syncthreads();
    }

    // --- epilogue: bias + store (C layout: M = r + 8*lh, N = l16) ---
    #pragma unroll
    for (int mt = 0; mt < 2; ++mt) {
        #pragma unroll
        for (int nt = 0; nt < 2; ++nt) {
            int gc = n0 + wn * 32 + nt * 16 + l16;
            float bv = bias[gc];
            #pragma unroll
            for (int r = 0; r < 8; ++r) {
                int gr = m0 + wm * 32 + mt * 16 + r + 8 * lh;
                float val = c[mt][nt][r] + bv;
                if (OUT_BF16)
                    ((unsigned short*)Cout)[(size_t)gr * N + gc] = f2bf(val);
                else
                    ((float*)Cout)[(size_t)gr * N + gc] = val;
            }
        }
    }
}

// ---------------------------------------------------------------------------
// Flash attention over grouped queries.
// QG: [B, S, G*H*dh] bf16 (row sg -> (s=sg/4, g=sg%4)); KP/VP: [B, S, D] bf16.
// Out: [B, SG, D] bf16. Grid: (SG/128 q-tiles, B*H). 8 waves, 16 q rows each.
// ---------------------------------------------------------------------------
__global__ __launch_bounds__(256) void flash_gqa(
    const unsigned short* __restrict__ QG, const unsigned short* __restrict__ KP,
    const unsigned short* __restrict__ VP, const int* __restrict__ maskp,
    unsigned short* __restrict__ Out)
{
    constexpr int S = 1024, D = 1024, dh = 64, SG = 4096, QT = 128, KB = 32;
    constexpr int VS = 40, PS = 40;
    __shared__ __align__(16) unsigned short Vts[64 * VS];        // V^T: [dh][keys]
    __shared__ __align__(16) unsigned short Ps[8][16 * PS];      // per-wave P block

    const int tid  = threadIdx.x;
    const int w    = tid >> 5;
    const int lane = tid & 31;
    const int l16  = lane & 15;
    const int lh   = lane >> 4;
    const int qt   = blockIdx.x;
    const int bh   = blockIdx.y;
    const int b    = bh >> 4;
    const int h    = bh & 15;

    // --- Q fragments straight from global in A layout (kept in registers) ---
    Frag16 aq[2];
    {
        int sg = qt * QT + w * 16 + l16;
        int si = sg >> 2, g = sg & 3;
        const unsigned short* qp =
            QG + ((size_t)(b * S + si)) * (4 * D) + g * D + h * dh;
        int k0 = lh * 8;
        #pragma unroll
        for (int s = 0; s < 2; ++s) {
            aq[s].q[0] = *(const uint4*)(qp + s * 32 + k0);
            aq[s].q[1] = *(const uint4*)(qp + s * 32 + 16 + k0);
        }
    }

    float mrow[8], lrow[8];
    #pragma unroll
    for (int r = 0; r < 8; ++r) { mrow[r] = -3.0e38f; lrow[r] = 0.0f; }
    v8f acc[4];
    #pragma unroll
    for (int n = 0; n < 4; ++n) acc[n] = vzero8();

    const float scl = 0.125f;   // 1/sqrt(64)

    for (int kb = 0; kb < S; kb += KB) {
        __syncthreads();        // Vts reuse fence
        {   // stage V^T (32 keys x 64 dh -> Vts[dh][key])
            int key = tid >> 3;
            int d8  = (tid & 7) << 3;
            U4S8 u;
            u.u = *(const uint4*)(VP + ((size_t)(b * S + kb + key)) * D + h * dh + d8);
            #pragma unroll
            for (int j = 0; j < 8; ++j) Vts[(d8 + j) * VS + key] = u.s[j];
        }
        __syncthreads();

        // --- scores: 16 q-rows x 32 keys per wave; K frags from global ---
        v8f sc[2];
        #pragma unroll
        for (int nt = 0; nt < 2; ++nt) {
            sc[nt] = vzero8();
            int key = kb + nt * 16 + l16;
            #pragma unroll
            for (int s = 0; s < 2; ++s) {
                Frag16 fk;
                const unsigned short* kp =
                    KP + ((size_t)(b * S + key)) * D + h * dh + s * 32 + lh * 8;
                fk.q[0] = *(const uint4*)(kp);
                fk.q[1] = *(const uint4*)(kp + 16);
                sc[nt] = __builtin_amdgcn_wmma_f32_16x16x32_bf16(
                    false, aq[s].v, false, fk.v, (short)0, sc[nt], false, false);
            }
        }

        int mk0 = maskp[b * S + kb + l16];
        int mk1 = maskp[b * S + kb + 16 + l16];

        float s0[8], s1[8], rm[8];
        #pragma unroll
        for (int r = 0; r < 8; ++r) {
            s0[r] = mk0 ? sc[0][r] * scl : -1.0e30f;
            s1[r] = mk1 ? sc[1][r] * scl : -1.0e30f;
            rm[r] = fmaxf(s0[r], s1[r]);
        }
        #pragma unroll
        for (int r = 0; r < 8; ++r) {     // row-max across 16 lanes of each half
            rm[r] = fmaxf(rm[r], __shfl_xor(rm[r], 1, 32));
            rm[r] = fmaxf(rm[r], __shfl_xor(rm[r], 2, 32));
            rm[r] = fmaxf(rm[r], __shfl_xor(rm[r], 4, 32));
            rm[r] = fmaxf(rm[r], __shfl_xor(rm[r], 8, 32));
        }

        float p0[8], p1[8], alpha[8], rs[8];
        #pragma unroll
        for (int r = 0; r < 8; ++r) {
            float mn = fmaxf(mrow[r], rm[r]);
            alpha[r] = __expf(mrow[r] - mn);
            p0[r] = mk0 ? __expf(s0[r] - mn) : 0.0f;
            p1[r] = mk1 ? __expf(s1[r] - mn) : 0.0f;
            mrow[r] = mn;
            rs[r] = p0[r] + p1[r];
        }
        #pragma unroll
        for (int r = 0; r < 8; ++r) {
            rs[r] += __shfl_xor(rs[r], 1, 32);
            rs[r] += __shfl_xor(rs[r], 2, 32);
            rs[r] += __shfl_xor(rs[r], 4, 32);
            rs[r] += __shfl_xor(rs[r], 8, 32);
            lrow[r] = lrow[r] * alpha[r] + rs[r];
        }
        #pragma unroll
        for (int n = 0; n < 4; ++n)
            #pragma unroll
            for (int r = 0; r < 8; ++r)
                acc[n][r] = acc[n][r] * alpha[r];

        // --- C-layout -> A-layout for P via per-wave LDS region ---
        #pragma unroll
        for (int r = 0; r < 8; ++r) {
            int row = r + 8 * lh;
            Ps[w][row * PS + l16]      = f2bf(p0[r]);
            Ps[w][row * PS + 16 + l16] = f2bf(p1[r]);
        }
        __syncthreads();

        Frag16 pa;
        {
            const unsigned short* pp = &Ps[w][l16 * PS];
            int k0 = lh * 8;
            pa.q[0] = *(const uint4*)(pp + k0);
            pa.q[1] = *(const uint4*)(pp + 16 + k0);
        }
        #pragma unroll
        for (int nt = 0; nt < 4; ++nt) {      // acc += P (16x32) x V (32x64)
            Frag16 fv;
            const unsigned short* vp = &Vts[(nt * 16 + l16) * VS + lh * 8];
            fv.q[0] = *(const uint4*)(vp);
            fv.q[1] = *(const uint4*)(vp + 16);
            acc[nt] = __builtin_amdgcn_wmma_f32_16x16x32_bf16(
                false, pa.v, false, fv.v, (short)0, acc[nt], false, false);
        }
    }

    // --- normalize and store bf16 to [B, SG, D] ---
    #pragma unroll
    for (int nt = 0; nt < 4; ++nt) {
        #pragma unroll
        for (int r = 0; r < 8; ++r) {
            int qrow = qt * QT + w * 16 + r + 8 * lh;
            int col  = h * dh + nt * 16 + l16;
            Out[((size_t)(b * SG + qrow)) * D + col] = f2bf(acc[nt][r] / lrow[r]);
        }
    }
}

// ---------------------------------------------------------------------------
// Host-side orchestration.
// ---------------------------------------------------------------------------
extern "C" void kernel_launch(void* const* d_in, const int* in_sizes, int n_in,
                              void* d_out, int out_size, void* d_ws, size_t ws_size,
                              hipStream_t stream)
{
    const float* q    = (const float*)d_in[0];
    const float* k    = (const float*)d_in[1];
    const float* v    = (const float*)d_in[2];
    const int*   mask = (const int*)  d_in[3];
    const float* Wq   = (const float*)d_in[4];
    const float* bq   = (const float*)d_in[5];
    const float* Wk   = (const float*)d_in[6];
    const float* bk   = (const float*)d_in[7];
    const float* Wv   = (const float*)d_in[8];
    const float* bv   = (const float*)d_in[9];
    const float* Wg   = (const float*)d_in[10];
    const float* bg   = (const float*)d_in[11];
    const float* Wo   = (const float*)d_in[12];
    const float* bo   = (const float*)d_in[13];

    (void)in_sizes; (void)n_in; (void)out_size; (void)ws_size;

    char* ws = (char*)d_ws;
    const size_t MB = 1u << 20;
    unsigned short* qb   = (unsigned short*)(ws + 0 * MB);   //  4 MiB [2048,1024]
    unsigned short* kb_  = (unsigned short*)(ws + 4 * MB);   //  4 MiB
    unsigned short* vb_  = (unsigned short*)(ws + 8 * MB);   //  4 MiB
    unsigned short* WqT  = (unsigned short*)(ws + 12 * MB);  //  2 MiB [1024,1024]
    unsigned short* WkT  = (unsigned short*)(ws + 14 * MB);  //  2 MiB
    unsigned short* WvT  = (unsigned short*)(ws + 16 * MB);  //  2 MiB
    unsigned short* WgT  = (unsigned short*)(ws + 18 * MB);  //  8 MiB [4096,1024]
    unsigned short* WoT  = (unsigned short*)(ws + 26 * MB);  //  2 MiB
    unsigned short* qp   = (unsigned short*)(ws + 28 * MB);  //  4 MiB [2048,1024]
    unsigned short* kp   = (unsigned short*)(ws + 32 * MB);  //  4 MiB
    unsigned short* vp   = (unsigned short*)(ws + 36 * MB);  //  4 MiB
    unsigned short* qg   = (unsigned short*)(ws + 40 * MB);  // 16 MiB [2048,4096]
    unsigned short* attn = (unsigned short*)(ws + 56 * MB);  // 16 MiB [8192,1024]

    dim3 blk(256);
    // one-time conversions: activations bf16, weights bf16 transposed [N][K]
    cvt_bf16<<<2048, blk, 0, stream>>>(q, qb);
    cvt_bf16<<<2048, blk, 0, stream>>>(k, kb_);
    cvt_bf16<<<2048, blk, 0, stream>>>(v, vb_);
    transpose_cvt_bf16<<<dim3(32, 32),  blk, 0, stream>>>(Wq, WqT, 1024, 1024);
    transpose_cvt_bf16<<<dim3(32, 32),  blk, 0, stream>>>(Wk, WkT, 1024, 1024);
    transpose_cvt_bf16<<<dim3(32, 32),  blk, 0, stream>>>(Wv, WvT, 1024, 1024);
    transpose_cvt_bf16<<<dim3(128, 32), blk, 0, stream>>>(Wg, WgT, 1024, 4096);
    transpose_cvt_bf16<<<dim3(32, 32),  blk, 0, stream>>>(Wo, WoT, 1024, 1024);

    // q/k/v projections (bf16 out)
    gemm_wmma<true><<<dim3(1024 / 64, 2048 / 128), blk, 0, stream>>>(qb,  WqT, bq, qp, 2048, 1024, 1024);
    gemm_wmma<true><<<dim3(1024 / 64, 2048 / 128), blk, 0, stream>>>(kb_, WkT, bk, kp, 2048, 1024, 1024);
    gemm_wmma<true><<<dim3(1024 / 64, 2048 / 128), blk, 0, stream>>>(vb_, WvT, bv, vp, 2048, 1024, 1024);
    // group expansion: qg = qp @ Wg + bg   [2048, 4096]
    gemm_wmma<true><<<dim3(4096 / 64, 2048 / 128), blk, 0, stream>>>(qp, WgT, bg, qg, 2048, 4096, 1024);
    // flash attention: 32 q-tiles x (B*H = 32)
    flash_gqa<<<dim3(32, 32), blk, 0, stream>>>(qg, kp, vp, mask, attn);
    // output projection (f32 out)
    gemm_wmma<false><<<dim3(1024 / 64, 8192 / 128), blk, 0, stream>>>(attn, WoT, bo, (float*)d_out, 8192, 1024, 1024);
}
